// GAT_29643864277421
// MI455X (gfx1250) — matrix operand order repaired and checked by reading, not compiled, not run on hardware
//
#include <hip/hip_runtime.h>
#include <hip/hip_bf16.h>
#include <stdint.h>

#define HID 128
#define SLOPE 0.2f

typedef __attribute__((ext_vector_type(16))) __bf16 v16bf;
typedef __attribute__((ext_vector_type(8)))  float  v8f;

union BF16Frag { v16bf v; unsigned short u[16]; };

__device__ __forceinline__ unsigned short f2bf(float f) {
  unsigned u = __float_as_uint(f);
  unsigned r = u + 0x7FFFu + ((u >> 16) & 1u);   // round-to-nearest-even
  return (unsigned short)(r >> 16);
}

// order-preserving float <-> uint encoding for atomicMax-based segment max
__device__ __forceinline__ unsigned enc_ord(float f) {
  unsigned u = __float_as_uint(f);
  return (u & 0x80000000u) ? ~u : (u | 0x80000000u);
}
__device__ __forceinline__ float dec_ord(unsigned u) {
  return (u & 0x80000000u) ? __uint_as_float(u & 0x7FFFFFFFu)
                           : __uint_as_float(~u);
}

// ---------------------------------------------------------------------------
// Weight pre-convert: f32 row-major [k][n] -> bf16 transposed [n][k]
// ---------------------------------------------------------------------------
__global__ void wconv_kernel(const float* __restrict__ W,
                             unsigned short* __restrict__ WT) {
  int idx = blockIdx.x * 256 + threadIdx.x;     // 0..16383
  int k = idx >> 7, n = idx & 127;
  WT[n * 128 + k] = f2bf(W[k * 128 + n]);
}

// ---------------------------------------------------------------------------
// Dual GEMM + bias:  XL = X@Wl + bl,  XR = X@Wr + br   (K = 128, bf16 WMMA)
// Block = 256 threads = 8 waves; block tile = 64 rows x 128 cols x 2 outputs.
// wave -> (output o = wave/4, mtile = wave%4); each wave: 8 ntiles of 16.
// Fragments loaded directly from global (L2-resident weights / activations).
// ---------------------------------------------------------------------------
__global__ __launch_bounds__(256)
void gemm_dual_wmma(const float* __restrict__ X,
                    const unsigned short* __restrict__ WlT,
                    const unsigned short* __restrict__ WrT,
                    const float* __restrict__ bl,
                    const float* __restrict__ br,
                    float* __restrict__ XL,
                    float* __restrict__ XR, int N) {
  const int wave = threadIdx.x >> 5;
  const int lane = threadIdx.x & 31;
  const int hi   = lane >> 4;       // half-wave select
  const int lo   = lane & 15;
  const int o     = wave >> 2;      // 0 -> Wl/XL, 1 -> Wr/XR
  const int mtile = wave & 3;
  const int rowbase = blockIdx.x * 64 + mtile * 16;
  const int arow    = rowbase + lo;

  const unsigned short* WT  = o ? WrT : WlT;
  const float*          bia = o ? br  : bl;
  float*                OUT = o ? XR  : XL;

  v8f acc[8];
  #pragma unroll
  for (int n = 0; n < 8; n++)
    #pragma unroll
    for (int r = 0; r < 8; r++) acc[n][r] = 0.f;

  #pragma unroll
  for (int kc = 0; kc < 4; kc++) {
    const int kbase = kc * 32;
    // ---- A fragment: 16x32 bf16. lane holds row `arow`;
    //      u[0..7]  = K = kbase + hi*8 + 0..7
    //      u[8..15] = K = kbase + 16 + hi*8 + 0..7
    BF16Frag A;
    if (arow < N) {
      const float* xp = X + (size_t)arow * 128;
      #pragma unroll
      for (int rr = 0; rr < 2; rr++) {
        const int k0 = kbase + rr * 16 + hi * 8;
        float4 f0 = *(const float4*)(xp + k0);
        float4 f1 = *(const float4*)(xp + k0 + 4);
        const int b = rr * 8;
        A.u[b + 0] = f2bf(f0.x); A.u[b + 1] = f2bf(f0.y);
        A.u[b + 2] = f2bf(f0.z); A.u[b + 3] = f2bf(f0.w);
        A.u[b + 4] = f2bf(f1.x); A.u[b + 5] = f2bf(f1.y);
        A.u[b + 6] = f2bf(f1.z); A.u[b + 7] = f2bf(f1.w);
      }
    } else {
      #pragma unroll
      for (int i = 0; i < 16; i++) A.u[i] = 0;
    }

    // ---- B fragments: 32x16 bf16 per ntile. lane holds col = ntile*16+lo;
    //      u[0..7] = WT[col][kbase + hi*8 + 0..7], u[8..15] = WT[col][kbase + 16 + hi*8 + 0..7]
    #pragma unroll
    for (int n = 0; n < 8; n++) {
      const int col = n * 16 + lo;
      const unsigned short* wp = WT + (size_t)col * 128 + kbase + hi * 8;
      BF16Frag B;
      *(uint4*)&B.u[0] = *(const uint4*)(wp);
      *(uint4*)&B.u[8] = *(const uint4*)(wp + 16);
      acc[n] = __builtin_amdgcn_wmma_f32_16x16x32_bf16(
          false, A.v, false, B.v, (short)0, acc[n], false, false);
    }
  }

  // ---- store D (16x16 f32: lane lo = col, VGPR r -> m = r + 8*hi) + bias
  #pragma unroll
  for (int n = 0; n < 8; n++) {
    const int col = n * 16 + lo;
    const float b = bia[col];
    #pragma unroll
    for (int r = 0; r < 8; r++) {
      const int orow = rowbase + r + 8 * hi;
      if (orow < N) OUT[(size_t)orow * 128 + col] = acc[n][r] + b;
    }
  }
}

// ---------------------------------------------------------------------------
// Per-layer init: segment max/sum and message accumulator
// ---------------------------------------------------------------------------
__global__ void init_layer_kernel(unsigned* __restrict__ mmax,
                                  float* __restrict__ ssum,
                                  float* __restrict__ acc, int N) {
  size_t tid = (size_t)blockIdx.x * 256 + threadIdx.x;
  if (tid < (size_t)N) { mmax[tid] = 0u; ssum[tid] = 0.f; }
  if (tid < (size_t)N * 128) acc[tid] = 0.f;
}

// ---------------------------------------------------------------------------
// Edge pass 1 (wave per edge): e = att . LeakyReLU(xl[src] + xr[dst])
// store e; atomicMax per-destination max (ordered-uint encoding)
// ---------------------------------------------------------------------------
__global__ __launch_bounds__(256)
void edge_max_kernel(const int* __restrict__ ei, int E0, int Etot,
                     const float* __restrict__ xl, const float* __restrict__ xr,
                     const float* __restrict__ att,
                     float* __restrict__ e_out, unsigned* __restrict__ mmax) {
  const int wid  = (int)(((size_t)blockIdx.x * 256 + threadIdx.x) >> 5);
  const int lane = threadIdx.x & 31;
  if (wid >= Etot) return;
  int s, d;
  if (wid < E0) { s = ei[wid]; d = ei[E0 + wid]; }
  else          { s = d = wid - E0; }

  const float4 a  = ((const float4*)att)[lane];
  const float4 va = ((const float4*)(xl + (size_t)s * 128))[lane];
  const float4 vb = ((const float4*)(xr + (size_t)d * 128))[lane];
  float t, p = 0.f;
  t = va.x + vb.x; p += a.x * (t > 0.f ? t : SLOPE * t);
  t = va.y + vb.y; p += a.y * (t > 0.f ? t : SLOPE * t);
  t = va.z + vb.z; p += a.z * (t > 0.f ? t : SLOPE * t);
  t = va.w + vb.w; p += a.w * (t > 0.f ? t : SLOPE * t);
  #pragma unroll
  for (int off = 16; off > 0; off >>= 1) p += __shfl_xor(p, off, 32);
  if (lane == 0) {
    e_out[wid] = p;
    atomicMax(&mmax[d], enc_ord(p));
  }
}

// ---------------------------------------------------------------------------
// Edge pass 2 (wave per edge): p = exp(e - max[dst]);
//   ssum[dst] += p ; acc[dst][:] += p * xl[src][:]
// (normalization p/ssum folded into node finalize)
// ---------------------------------------------------------------------------
__global__ __launch_bounds__(256)
void edge_acc_kernel(const int* __restrict__ ei, int E0, int Etot,
                     const float* __restrict__ xl,
                     const float* __restrict__ e_in,
                     const unsigned* __restrict__ mmax,
                     float* __restrict__ ssum, float* __restrict__ acc) {
  const int wid  = (int)(((size_t)blockIdx.x * 256 + threadIdx.x) >> 5);
  const int lane = threadIdx.x & 31;
  if (wid >= Etot) return;
  int s, d;
  if (wid < E0) { s = ei[wid]; d = ei[E0 + wid]; }
  else          { s = d = wid - E0; }

  const float p = __expf(e_in[wid] - dec_ord(mmax[d]));
  if (lane == 0) atomicAdd(&ssum[d], p);
  const float4 v = ((const float4*)(xl + (size_t)s * 128))[lane];
  float* ap = acc + (size_t)d * 128 + lane * 4;
  atomicAdd(ap + 0, p * v.x);
  atomicAdd(ap + 1, p * v.y);
  atomicAdd(ap + 2, p * v.z);
  atomicAdd(ap + 3, p * v.w);
}

// ---------------------------------------------------------------------------
// Node finalize: h = acc / ssum + bias ; optional ReLU   (in-place safe)
// ---------------------------------------------------------------------------
__global__ void node_fin_kernel(const float* __restrict__ acc,
                                const float* __restrict__ ssum,
                                const float* __restrict__ bias,
                                float* __restrict__ h, int N, int do_relu) {
  size_t tid = (size_t)blockIdx.x * 256 + threadIdx.x;
  if (tid >= (size_t)N * 128) return;
  int i = (int)(tid >> 7), f = (int)(tid & 127);
  float v = acc[tid] / ssum[i] + bias[f];
  if (do_relu) v = v > 0.f ? v : 0.f;
  h[tid] = v;
}

// ---------------------------------------------------------------------------
// Mean pooling over graph ids
// ---------------------------------------------------------------------------
__global__ void init_pool_kernel(float* __restrict__ sums,
                                 float* __restrict__ cnt) {
  int tid = blockIdx.x * 256 + threadIdx.x;
  if (tid < 512 * 128) sums[tid] = 0.f;
  if (tid < 512) cnt[tid] = 0.f;
}

__global__ void pool_acc_kernel(const float* __restrict__ h,
                                const int* __restrict__ batch,
                                float* __restrict__ sums,
                                float* __restrict__ cnt, int N) {
  size_t tid = (size_t)blockIdx.x * 256 + threadIdx.x;
  if (tid >= (size_t)N * 128) return;
  int i = (int)(tid >> 7), f = (int)(tid & 127);
  int b = batch[i];
  atomicAdd(&sums[(size_t)b * 128 + f], h[tid]);
  if (f == 0) atomicAdd(&cnt[b], 1.f);
}

// ---------------------------------------------------------------------------
// MLP head: g = sums/max(cnt,1); sigmoid(g@W4+b4); sigmoid(@W5+b5); @W6+b6
// one block (128 threads) per graph
// ---------------------------------------------------------------------------
__global__ __launch_bounds__(128)
void head_kernel(const float* __restrict__ sums, const float* __restrict__ cnt,
                 const float* __restrict__ W4, const float* __restrict__ b4,
                 const float* __restrict__ W5, const float* __restrict__ b5,
                 const float* __restrict__ W6, const float* __restrict__ b6,
                 float* __restrict__ out) {
  __shared__ float g[128], h1[128], h2[64];
  const int b = blockIdx.x, t = threadIdx.x;
  float c = cnt[b]; c = c > 1.f ? c : 1.f;
  g[t] = sums[(size_t)b * 128 + t] / c;
  __syncthreads();
  float s = b4[t];
  for (int k = 0; k < 128; k++) s += g[k] * W4[k * 128 + t];
  h1[t] = 1.f / (1.f + __expf(-s));
  __syncthreads();
  if (t < 64) {
    float s2 = b5[t];
    for (int k = 0; k < 128; k++) s2 += h1[k] * W5[k * 64 + t];
    h2[t] = 1.f / (1.f + __expf(-s2));
  }
  __syncthreads();
  if (t < 2) {
    float s3 = b6[t];
    for (int k = 0; k < 64; k++) s3 += h2[k] * W6[k * 2 + t];
    out[b * 2 + t] = s3;
  }
}

// ---------------------------------------------------------------------------
extern "C" void kernel_launch(void* const* d_in, const int* in_sizes, int n_in,
                              void* d_out, int out_size, void* d_ws,
                              size_t ws_size, hipStream_t stream) {
  const float* x     = (const float*)d_in[0];
  const int*   ei    = (const int*)d_in[1];
  const int*   batch = (const int*)d_in[2];
  const float *Wl[3], *bl[3], *Wr[3], *br[3], *att[3], *bias[3];
  for (int L = 0; L < 3; L++) {
    int base = 3 + 6 * L;
    Wl[L]   = (const float*)d_in[base + 0];
    bl[L]   = (const float*)d_in[base + 1];
    Wr[L]   = (const float*)d_in[base + 2];
    br[L]   = (const float*)d_in[base + 3];
    att[L]  = (const float*)d_in[base + 4];
    bias[L] = (const float*)d_in[base + 5];
  }
  const float* W4 = (const float*)d_in[21]; const float* b4 = (const float*)d_in[22];
  const float* W5 = (const float*)d_in[23]; const float* b5 = (const float*)d_in[24];
  const float* W6 = (const float*)d_in[25]; const float* b6 = (const float*)d_in[26];

  const int N    = in_sizes[0] / 128;
  const int E0   = in_sizes[1] / 2;
  const int Etot = E0 + N;

  // ---- workspace carve (256B aligned regions) ----
  char* w = (char*)d_ws;
  auto carve = [&](size_t bytes) {
    void* p = (void*)w;
    w += (bytes + 255) & ~(size_t)255;
    return p;
  };
  const size_t nodeB = (size_t)N * 128 * sizeof(float);
  float*    xl_   = (float*)carve(nodeB);
  float*    xr_   = (float*)carve(nodeB);
  float*    hbuf  = (float*)carve(nodeB);     // doubles as acc + layer output
  float*    e_    = (float*)carve((size_t)Etot * sizeof(float));
  unsigned* mmax  = (unsigned*)carve((size_t)N * sizeof(unsigned));
  float*    ssum  = (float*)carve((size_t)N * sizeof(float));
  unsigned short* wbfL[3]; unsigned short* wbfR[3];
  for (int L = 0; L < 3; L++) {
    wbfL[L] = (unsigned short*)carve(128 * 128 * sizeof(unsigned short));
    wbfR[L] = (unsigned short*)carve(128 * 128 * sizeof(unsigned short));
  }
  float* psums = (float*)carve(512 * 128 * sizeof(float));
  float* pcnt  = (float*)carve(512 * sizeof(float));

  // ---- weight conversion to bf16 (transposed [n][k]) ----
  for (int L = 0; L < 3; L++) {
    wconv_kernel<<<64, 256, 0, stream>>>(Wl[L], wbfL[L]);
    wconv_kernel<<<64, 256, 0, stream>>>(Wr[L], wbfR[L]);
  }

  const unsigned gNodeF = (unsigned)(((size_t)N * 128 + 255) / 256);
  const unsigned gGemm  = (unsigned)((N + 63) / 64);
  const unsigned gEdge  = (unsigned)((Etot + 7) / 8);

  const float* hin = x;
  for (int L = 0; L < 3; L++) {
    gemm_dual_wmma<<<gGemm, 256, 0, stream>>>(hin, wbfL[L], wbfR[L], bl[L],
                                              br[L], xl_, xr_, N);
    init_layer_kernel<<<gNodeF, 256, 0, stream>>>(mmax, ssum, hbuf, N);
    edge_max_kernel<<<gEdge, 256, 0, stream>>>(ei, E0, Etot, xl_, xr_, att[L],
                                               e_, mmax);
    edge_acc_kernel<<<gEdge, 256, 0, stream>>>(ei, E0, Etot, xl_, e_, mmax,
                                               ssum, hbuf);
    node_fin_kernel<<<gNodeF, 256, 0, stream>>>(hbuf, ssum, bias[L], hbuf, N,
                                                L < 2 ? 1 : 0);
    hin = hbuf;
  }

  init_pool_kernel<<<256, 256, 0, stream>>>(psums, pcnt);
  pool_acc_kernel<<<gNodeF, 256, 0, stream>>>(hbuf, batch, psums, pcnt, N);
  head_kernel<<<512, 128, 0, stream>>>(psums, pcnt, W4, b4, W5, b5, W6, b6,
                                       (float*)d_out);
}
